// Top_Down_With_Pair_Rf_66692252172737
// MI455X (gfx1250) — compile-verified, matrix operand order checked by reading, not run on hardware
//
#include <hip/hip_runtime.h>
#include <cstddef>

// ---------------------------------------------------------------- constants
#define B_    256
#define K_    49
#define C_    2048
#define R_    6
#define E_    300
#define H_    2048
#define A_    512
#define L_    10000
#define BR_   (B_ * R_)       // 1536
#define BKR_  (B_ * K_)       // 12544
#define E2_   (2 * E_)        // 600
#define E2P_  608             // 600 padded to 32-multiple (K dim of GEMM 1)
#define LP_   10112           // 10000 padded to 128-multiple (N dim of GEMM 12)

// ---------------------------------------------------------------- vector types
typedef __attribute__((ext_vector_type(16))) _Float16 v16h;
typedef __attribute__((ext_vector_type(8)))  _Float16 v8h;
typedef __attribute__((ext_vector_type(8)))  float    v8f;
typedef int iv4 __attribute__((vector_size(16)));   // matches async builtin param

#define AS3 __attribute__((address_space(3)))

#if defined(__gfx1250__) && __has_builtin(__builtin_amdgcn_global_load_async_to_lds_b128)
#define HAVE_ASYNC 1
#else
#define HAVE_ASYNC 0
#endif

// ---------------------------------------------------------------- WMMA GEMM
// D(M,Nreal) = A(M,Kp) @ Bt(Np,Kp)^T [+bias] [+relu].
// A row-major M x Kp f16, Bt row-major Np x Kp f16 (pre-transposed weights),
// both zero-padded so loads need no bounds checks. M%128==0, Kp%32==0,
// Np%128==0; stores masked by col < Nreal. Double-buffered async LDS staging.
#define BM  128
#define BN  128
#define BKT 32
#define LDST 40   // 32 + 8 halves pad: 80B row stride, 16B aligned, conflict-free

// 16x32 f16 fragment (rows contiguous in LDS, stride LDST halves):
// lane 0-15 -> rows, K 0..7 / 16..23 ; lane 16-31 -> rows, K 8..15 / 24..31
__device__ __forceinline__ v16h frag_ld(const _Float16* rowp, int half) {
  v8h lo = *(const v8h*)(rowp + half * 8);
  v8h hi = *(const v8h*)(rowp + 16 + half * 8);
  v16h o;
#pragma unroll
  for (int i = 0; i < 8; ++i) { o[i] = lo[i]; o[i + 8] = hi[i]; }
  return o;
}

// EPI: 0 = none, 1 = +bias, 2 = +bias then relu.  HOUT: store f16 instead of f32.
template <int EPI, bool HOUT>
__global__ __launch_bounds__(256) void gemm_wmma(
    const _Float16* __restrict__ A, const _Float16* __restrict__ Bt,
    const float* __restrict__ bias, void* __restrict__ Dv,
    int M, int Nreal, int Kp) {
  __shared__ __align__(16) _Float16 As[2][BM * LDST];
  __shared__ __align__(16) _Float16 Bs[2][BN * LDST];

  const int tid  = threadIdx.x;
  const int lane = tid & 31;
  const int wid  = tid >> 5;     // 0..7
  const int wm   = wid & 1;      // 2 strips of 64 rows
  const int wn   = wid >> 1;     // 4 strips of 32 cols
  const int m0   = blockIdx.y * BM;
  const int n0   = blockIdx.x * BN;
  const int half = lane >> 4;
  const int r    = lane & 15;

  v8f acc[4][2];
#pragma unroll
  for (int i = 0; i < 4; ++i)
#pragma unroll
    for (int j = 0; j < 2; ++j) acc[i][j] = (v8f){0, 0, 0, 0, 0, 0, 0, 0};

  // stage one 128x32 A tile + one 128x32 B tile: 4 x 16B async copies per lane
  auto stage = [&](int bi, int kt) {
#pragma unroll
    for (int it = 0; it < 2; ++it) {
      const int id  = tid + it * 256;   // 0..511
      const int row = id >> 2;          // 0..127
      const int off = (id & 3) * 8;     // halves
      const _Float16* ga = A  + (size_t)(m0 + row) * Kp + kt + off;
      const _Float16* gb = Bt + (size_t)(n0 + row) * Kp + kt + off;
      _Float16* la = &As[bi][row * LDST + off];
      _Float16* lb = &Bs[bi][row * LDST + off];
#if HAVE_ASYNC
      __builtin_amdgcn_global_load_async_to_lds_b128((iv4*)ga, (AS3 iv4*)la, 0, 0);
      __builtin_amdgcn_global_load_async_to_lds_b128((iv4*)gb, (AS3 iv4*)lb, 0, 0);
#else
      *(v8h*)la = *(const v8h*)ga;
      *(v8h*)lb = *(const v8h*)gb;
#endif
    }
  };

  const int nk = Kp / BKT;
  stage(0, 0);
  for (int i = 0; i < nk; ++i) {
    const int bi = i & 1;
    if (i + 1 < nk) {
      stage(bi ^ 1, (i + 1) * BKT);   // prefetch next tile into other buffer
#if HAVE_ASYNC
      asm volatile("s_wait_asynccnt 4" ::: "memory");  // oldest 4 (this tile) done
#endif
    } else {
#if HAVE_ASYNC
      asm volatile("s_wait_asynccnt 0" ::: "memory");
#endif
    }
    __syncthreads();

    v16h fa[4], fb[2];
#pragma unroll
    for (int x = 0; x < 4; ++x)
      fa[x] = frag_ld(&As[bi][(wm * 64 + x * 16 + r) * LDST], half);
#pragma unroll
    for (int j = 0; j < 2; ++j)
      fb[j] = frag_ld(&Bs[bi][(wn * 32 + j * 16 + r) * LDST], half);
#pragma unroll
    for (int x = 0; x < 4; ++x)
#pragma unroll
      for (int j = 0; j < 2; ++j)
        acc[x][j] = __builtin_amdgcn_wmma_f32_16x16x32_f16(
            false, fa[x], false, fb[j], (short)0, acc[x][j], false, false);
    __syncthreads();   // protect buffer bi before it is refilled at i+2
  }

  // ---- epilogue: per 16x16 tile, row = v + 8*(lane>>4), col = lane&15
  float*    Df = (float*)Dv;
  _Float16* Dh = (_Float16*)Dv;
#pragma unroll
  for (int x = 0; x < 4; ++x)
#pragma unroll
    for (int j = 0; j < 2; ++j) {
      const int col = n0 + wn * 32 + j * 16 + (lane & 15);
      const int rowBase = m0 + wm * 64 + x * 16 + (lane >> 4) * 8;
      if (col < Nreal) {
        const float bval = (EPI >= 1) ? bias[col] : 0.0f;
#pragma unroll
        for (int v = 0; v < 8; ++v) {
          float val = acc[x][j][v] + bval;
          if (EPI == 2) val = fmaxf(val, 0.0f);
          const size_t idx = (size_t)(rowBase + v) * Nreal + col;
          if (HOUT) Dh[idx] = (_Float16)val;
          else      Df[idx] = val;
        }
      }
    }
}

// ------------------------------------------------- f32 -> f16 convert (row-major, pad)
__global__ void cvt_f16_pad(const float* __restrict__ src, _Float16* __restrict__ dst,
                            int Ms, int Ns, int Md, int Nd) {
  int i = blockIdx.x * 256 + threadIdx.x;
  if (i >= Md * Nd) return;
  int rr = i / Nd, c = i - rr * Nd;
  float v = (rr < Ms && c < Ns) ? src[(size_t)rr * Ns + c] : 0.0f;
  dst[i] = (_Float16)v;
}

// ------------------------------------------------- f32 -> f16 transpose w/ pad
// src: Ks x Ns row-major f32; dst: Nd x Kd row-major f16 (dst[n][k] = src[k][n],
// zero outside). Coalesced both sides via 32x33 LDS tile.
__global__ __launch_bounds__(256) void cvt_f16_T(
    const float* __restrict__ src, _Float16* __restrict__ dst,
    int Ks, int Ns, int Kd, int Nd) {
  __shared__ float t[32][33];
  const int tk0 = blockIdx.y * 32, tn0 = blockIdx.x * 32;
  const int lr = threadIdx.x >> 5;   // 0..7
  const int lc = threadIdx.x & 31;
#pragma unroll
  for (int s = 0; s < 32; s += 8) {
    int k = tk0 + lr + s, n = tn0 + lc;
    t[lr + s][lc] = (k < Ks && n < Ns) ? src[(size_t)k * Ns + n] : 0.0f;
  }
  __syncthreads();
#pragma unroll
  for (int s = 0; s < 32; s += 8) {
    int n = tn0 + lr + s, k = tk0 + lc;
    if (n < Nd && k < Kd) dst[(size_t)n * Kd + k] = (_Float16)t[lc][lr + s];
  }
}

// ---------------------------------------------------------------- q_in gather (f16, K-padded)
__global__ void build_qin(const int* __restrict__ gt_verb,
                          const int* __restrict__ role_idx,
                          const float* __restrict__ verb_table,
                          const float* __restrict__ role_table,
                          _Float16* __restrict__ q_in) {
  int idx = blockIdx.x * 256 + threadIdx.x;
  if (idx >= BR_ * E2P_) return;
  int row = idx / E2P_, col = idx - row * E2P_;
  int b = row / R_, rr = row - b * R_;
  float v = 0.0f;
  if (col < E_)        v = verb_table[(size_t)gt_verb[b] * E_ + col];
  else if (col < E2_)  v = role_table[(size_t)role_idx[b * R_ + rr] * E_ + (col - E_)];
  q_in[idx] = (_Float16)v;
}

// ---------------------------------------------------------------- verb attention
__global__ __launch_bounds__(256) void verb_attn(
    const float* __restrict__ vProj,      // [B*K, A]
    const float* __restrict__ qProj,      // [B*R, A]
    const float* __restrict__ va_w,       // [A]
    const _Float16* __restrict__ v_orgh,  // [B, K, C] f16
    _Float16* __restrict__ v_embh) {      // [B*R, C] f16
  const int br = blockIdx.x;
  const int b  = br / R_;
  __shared__ float sc[K_];
  const int tid = threadIdx.x, lane = tid & 31, wid = tid >> 5;
  const float* qp = qProj + (size_t)br * A_;

  for (int k = wid; k < K_; k += 8) {
    const float* vp = vProj + (size_t)(b * K_ + k) * A_;
    float p = 0.0f;
    for (int a = lane; a < A_; a += 32)
      p += fmaxf(vp[a] + qp[a], 0.0f) * va_w[a];
#pragma unroll
    for (int off = 16; off >= 1; off >>= 1) p += __shfl_xor(p, off, 32);
    if (lane == 0) sc[k] = p;
  }
  __syncthreads();
  if (tid == 0) {
    float m = sc[0];
    for (int k = 1; k < K_; ++k) m = fmaxf(m, sc[k]);
    float s = 0.0f;
    for (int k = 0; k < K_; ++k) { float e = __expf(sc[k] - m); sc[k] = e; s += e; }
    float inv = 1.0f / s;
    for (int k = 0; k < K_; ++k) sc[k] *= inv;
  }
  __syncthreads();
  const _Float16* vo = v_orgh + (size_t)b * K_ * C_;
  _Float16* ve = v_embh + (size_t)br * C_;
  for (int c = tid; c < C_; c += 256) {
    float acc = 0.0f;
#pragma unroll 7
    for (int k = 0; k < K_; ++k) acc += sc[k] * (float)vo[(size_t)k * C_ + c];
    ve[c] = (_Float16)acc;
  }
}

// ------------------------------------------- all_roles = a*b -> f32 and f16 copies
__global__ void ew_mul_dual(const float* __restrict__ a, const float* __restrict__ b,
                            float* __restrict__ o, _Float16* __restrict__ oh, int n) {
  int i = blockIdx.x * 256 + threadIdx.x;
  if (i < n) { float v = a[i] * b[i]; o[i] = v; oh[i] = (_Float16)v; }
}

// ---------------------------------------------------------------- pair attention
__global__ __launch_bounds__(256) void pair_attn(
    const float* __restrict__ nProj, const float* __restrict__ cProj,
    const float* __restrict__ na_w, const float* __restrict__ all_roles,
    _Float16* __restrict__ jointh) {
  const int br = blockIdx.x;
  const int b = br / R_, rr = br - b * R_;
  __shared__ float sw[R_ - 1];
  const int tid = threadIdx.x, lane = tid & 31, wid = tid >> 5;
  const float* cp = cProj + (size_t)br * A_;

  if (wid < R_ - 1) {
    const int j  = wid;
    const int nr = (j < rr) ? j : j + 1;
    const float* np = nProj + (size_t)(b * R_ + nr) * A_;
    float p = 0.0f;
    for (int a = lane; a < A_; a += 32)
      p += fmaxf(np[a] + cp[a], 0.0f) * na_w[a];
#pragma unroll
    for (int off = 16; off >= 1; off >>= 1) p += __shfl_xor(p, off, 32);
    if (lane == 0) sw[j] = p;
  }
  __syncthreads();
  if (tid == 0) {
    float m = sw[0];
    for (int j = 1; j < R_ - 1; ++j) m = fmaxf(m, sw[j]);
    float s = 0.0f;
    for (int j = 0; j < R_ - 1; ++j) { float e = __expf(sw[j] - m); sw[j] = e; s += e; }
    float inv = 1.0f / s;
    for (int j = 0; j < R_ - 1; ++j) sw[j] *= inv;
  }
  __syncthreads();
  const float* ar_b = all_roles + (size_t)(b * R_) * H_;
  const float* ar_r = ar_b + (size_t)rr * H_;
  _Float16* jp = jointh + (size_t)br * H_;
  for (int c = tid; c < H_; c += 256) {
    float ctx = 0.0f;
#pragma unroll
    for (int j = 0; j < R_ - 1; ++j) {
      const int nr = (j < rr) ? j : j + 1;
      ctx += sw[j] * ar_b[(size_t)nr * H_ + c];
    }
    jp[c] = (_Float16)(ar_r[c] + ctx);
  }
}

// ---------------------------------------------------------------- launch
static inline dim3 gemm_grid(int M, int Np) { return dim3(Np / BN, M / BM); }
static inline dim3 tr_grid(int Kd, int Nd)  { return dim3((Nd + 31) / 32, (Kd + 31) / 32); }
static inline int  lin_blocks(size_t n)     { return (int)((n + 255) / 256); }

extern "C" void kernel_launch(void* const* d_in, const int* in_sizes, int n_in,
                              void* d_out, int out_size, void* d_ws, size_t ws_size,
                              hipStream_t stream) {
  const float* v_org      = (const float*)d_in[0];
  const int*   gt_verb    = (const int*)  d_in[1];
  const int*   role_idx   = (const int*)  d_in[2];
  const float* verb_table = (const float*)d_in[3];
  const float* role_table = (const float*)d_in[4];
  const float* qc_W  = (const float*)d_in[5];
  const float* qc_b  = (const float*)d_in[6];
  const float* va_Wv = (const float*)d_in[7];
  const float* va_Wq = (const float*)d_in[8];
  const float* va_w  = (const float*)d_in[9];
  const float* qn_W  = (const float*)d_in[10];
  const float* qn_b  = (const float*)d_in[11];
  const float* vn_W  = (const float*)d_in[12];
  const float* vn_b  = (const float*)d_in[13];
  const float* na_Wn = (const float*)d_in[14];
  const float* na_Wc = (const float*)d_in[15];
  const float* na_w  = (const float*)d_in[16];
  const float* cl_W  = (const float*)d_in[17];
  const float* cl_b  = (const float*)d_in[18];
  float* logits = (float*)d_out;

  // ---- workspace carve (256B aligned chunks)
  char* wp = (char*)d_ws;
  auto alloc = [&](size_t bytes) -> char* {
    char* p = wp; wp += (bytes + 255) & ~(size_t)255; return p;
  };
  _Float16* q_inh  = (_Float16*)alloc((size_t)BR_ * E2P_ * 2);
  _Float16* qembh  = (_Float16*)alloc((size_t)BR_ * H_ * 2);
  _Float16* vembh  = (_Float16*)alloc((size_t)BR_ * H_ * 2);
  _Float16* arh    = (_Float16*)alloc((size_t)BR_ * H_ * 2);
  _Float16* jointh = (_Float16*)alloc((size_t)BR_ * H_ * 2);
  _Float16* vorgh  = (_Float16*)alloc((size_t)BKR_ * C_ * 2);
  // weights stored TRANSPOSED (N x K) f16, zero-padded
  _Float16* qcWt  = (_Float16*)alloc((size_t)H_ * E2P_ * 2);   // 2048 x 608
  _Float16* vaWvt = (_Float16*)alloc((size_t)A_ * C_ * 2);     // 512 x 2048
  _Float16* vaWqt = (_Float16*)alloc((size_t)A_ * H_ * 2);
  _Float16* qnWt  = (_Float16*)alloc((size_t)H_ * H_ * 2);
  _Float16* vnWt  = (_Float16*)alloc((size_t)H_ * C_ * 2);
  _Float16* naWnt = (_Float16*)alloc((size_t)A_ * H_ * 2);
  _Float16* naWct = (_Float16*)alloc((size_t)A_ * H_ * 2);
  _Float16* clWt  = (_Float16*)alloc((size_t)LP_ * H_ * 2);    // 10112 x 2048
  float* vProj  = (float*)alloc((size_t)BKR_ * A_ * 4);        // later nProj/cProj
  float* qProj  = (float*)alloc((size_t)BR_ * A_ * 4);
  float* v_repr = (float*)alloc((size_t)BR_ * H_ * 4);         // later all_roles
  float* q_repr = (float*)alloc((size_t)BR_ * H_ * 4);
  float* nProj  = vProj;
  float* cProj  = vProj + (size_t)BR_ * A_;

  // ---- one-time (per call) conversions: A-side row-major, B-side transposed
  cvt_f16_pad<<<lin_blocks((size_t)BKR_ * C_), 256, 0, stream>>>(v_org, vorgh, BKR_, C_, BKR_, C_);
  cvt_f16_T<<<tr_grid(E2P_, H_), 256, 0, stream>>>(qc_W,  qcWt,  E2_, H_, E2P_, H_);
  cvt_f16_T<<<tr_grid(C_, A_),   256, 0, stream>>>(va_Wv, vaWvt, C_,  A_, C_,  A_);
  cvt_f16_T<<<tr_grid(H_, A_),   256, 0, stream>>>(va_Wq, vaWqt, H_,  A_, H_,  A_);
  cvt_f16_T<<<tr_grid(H_, H_),   256, 0, stream>>>(qn_W,  qnWt,  H_,  H_, H_,  H_);
  cvt_f16_T<<<tr_grid(C_, H_),   256, 0, stream>>>(vn_W,  vnWt,  C_,  H_, C_,  H_);
  cvt_f16_T<<<tr_grid(H_, A_),   256, 0, stream>>>(na_Wn, naWnt, H_,  A_, H_,  A_);
  cvt_f16_T<<<tr_grid(H_, A_),   256, 0, stream>>>(na_Wc, naWct, H_,  A_, H_,  A_);
  cvt_f16_T<<<tr_grid(H_, LP_),  256, 0, stream>>>(cl_W,  clWt,  H_,  L_, H_,  LP_);

  // 1. embeddings gather -> q_in f16 [1536, 608]
  build_qin<<<lin_blocks((size_t)BR_ * E2P_), 256, 0, stream>>>(
      gt_verb, role_idx, verb_table, role_table, q_inh);
  // 2. q_emb = relu(q_in @ qc_W + qc_b) -> f16 [1536, 2048]
  gemm_wmma<2, true><<<gemm_grid(BR_, H_), 256, 0, stream>>>(
      q_inh, qcWt, qc_b, qembh, BR_, H_, E2P_);
  // 3. vProj = v_org @ va_Wv [12544, 512]
  gemm_wmma<0, false><<<gemm_grid(BKR_, A_), 256, 0, stream>>>(
      vorgh, vaWvt, nullptr, vProj, BKR_, A_, C_);
  // 4. qProj = q_emb @ va_Wq [1536, 512]
  gemm_wmma<0, false><<<gemm_grid(BR_, A_), 256, 0, stream>>>(
      qembh, vaWqt, nullptr, qProj, BR_, A_, H_);
  // 5. verb attention -> v_emb f16 [1536, 2048]
  verb_attn<<<BR_, 256, 0, stream>>>(vProj, qProj, va_w, vorgh, vembh);
  // 6. v_repr = relu(v_emb @ vn_W + vn_b) [1536, 2048] f32
  gemm_wmma<2, false><<<gemm_grid(BR_, H_), 256, 0, stream>>>(
      vembh, vnWt, vn_b, v_repr, BR_, H_, C_);
  // 7. q_repr = relu(q_emb @ qn_W + qn_b) [1536, 2048] f32
  gemm_wmma<2, false><<<gemm_grid(BR_, H_), 256, 0, stream>>>(
      qembh, qnWt, qn_b, q_repr, BR_, H_, H_);
  // 8. all_roles = q_repr * v_repr -> f32 (in v_repr) + f16 (arh)
  ew_mul_dual<<<lin_blocks((size_t)BR_ * H_), 256, 0, stream>>>(
      q_repr, v_repr, v_repr, arh, BR_ * H_);
  // 9/10. nProj / cProj = all_roles @ na_Wn / na_Wc [1536, 512]
  gemm_wmma<0, false><<<gemm_grid(BR_, A_), 256, 0, stream>>>(
      arh, naWnt, nullptr, nProj, BR_, A_, H_);
  gemm_wmma<0, false><<<gemm_grid(BR_, A_), 256, 0, stream>>>(
      arh, naWct, nullptr, cProj, BR_, A_, H_);
  // 11. pair attention -> joint f16 [1536, 2048]
  pair_attn<<<BR_, 256, 0, stream>>>(nProj, cProj, na_w, v_repr, jointh);
  // 12. logits = joint @ cl_W + cl_b [1536, 10000] (Bt padded to 10112 rows)
  gemm_wmma<1, false><<<gemm_grid(BR_, LP_), 256, 0, stream>>>(
      jointh, clWt, cl_b, logits, BR_, L_, H_);

  (void)in_sizes; (void)n_in; (void)out_size; (void)ws_size;
}
